// ScaledDotProductAttention_42640435315317
// MI455X (gfx1250) — matrix operand order repaired and checked by reading, not compiled, not run on hardware
//
#include <hip/hip_runtime.h>

typedef __attribute__((ext_vector_type(16))) _Float16 v16h;
typedef __attribute__((ext_vector_type(8)))  float    v8f;

#define D_MODEL 1024
#define NHEAD   16
#define D_KH    64
#define NB      4
#define NQS     1024
#define NKS     1024
#define BH      (NB * NHEAD)
#define M_ALL   (NB * NQS)   // 4096

// ---- WMMA fragment helpers (f16, 16x16x32) --------------------------------
// A-matrix 16x32: lane<16 -> M=lane, K=0..15;
//                 lane>=16 -> M=lane-16, K=16..31.
__device__ __forceinline__ v16h frag_ldA(const _Float16* p, int ldk) {
  const int lane = threadIdx.x & 31;
  const int r  = lane & 15;
  const int k0 = (lane >> 4) << 4;
  const _Float16* q = p + (size_t)r * ldk + k0;
  v16h f;
#pragma unroll
  for (int i = 0; i < 8; ++i) f[i] = q[i];
#pragma unroll
  for (int i = 0; i < 8; ++i) f[8 + i] = q[8 + i];
  return f;
}

// B-matrix 32x16: lane group 0 -> K=0..15, lane group 1 -> K=16..31,
// column n = lane&15.  Source stored [n][k] (contiguous K per column).
__device__ __forceinline__ v16h frag_ldB(const _Float16* p, int ldk) {
  const int lane = threadIdx.x & 31;
  const _Float16* q = p + (size_t)(lane & 15) * ldk + ((lane >> 4) << 4);
  v16h f;
#pragma unroll
  for (int i = 0; i < 16; ++i) f[i] = q[i];
  return f;
}

__device__ __forceinline__ v8f wmma16(v16h a, v16h b, v8f c) {
  return __builtin_amdgcn_wmma_f32_16x16x32_f16(false, a, false, b,
                                                (short)0, c, false, false);
}

// ---- GEMM tiling ----------------------------------------------------------
#define KT   32
#define MT   128
#define NTT  128
#define PAD  8
#define LDT  (KT + PAD)
#define TBUF (MT * LDT)      // elems per LDS buffer (A and B tiles both 128x32)

// ---- Projection GEMM:  C = A(f32, MxK) @ W(f32, NxK)^T + bias -------------
// mode 0: write f16 head-split (b,h,n,64); mode 1: write f16 transposed (b,h,64,n)
__global__ __launch_bounds__(256) void proj_gemm_kernel(
    const float* __restrict__ A, const float* __restrict__ W,
    const float* __restrict__ bias, _Float16* __restrict__ out, int mode) {
  __shared__ _Float16 sA[2 * TBUF];
  __shared__ _Float16 sB[2 * TBUF];
  const int tid   = threadIdx.x;
  const int wid   = tid >> 5;
  const int lane  = tid & 31;
  const int mBase = blockIdx.y * MT;
  const int nBase = blockIdx.x * NTT;

  v8f acc[8];
#pragma unroll
  for (int t = 0; t < 8; ++t) acc[t] = (v8f){};

  float4 ra[4], rb[4];
  // prologue: stage k-step 0 into buffer 0
#pragma unroll
  for (int i = 0; i < 4; ++i) {
    const int chunk = tid + i * 256;        // 1024 float4 chunks per tile
    const int r = chunk >> 3;
    const int c = (chunk & 7) << 2;
    ra[i] = *(const float4*)(A + (size_t)(mBase + r) * D_MODEL + c);
    rb[i] = *(const float4*)(W + (size_t)(nBase + r) * D_MODEL + c);
  }
#pragma unroll
  for (int i = 0; i < 4; ++i) {
    const int chunk = tid + i * 256;
    const int r = chunk >> 3;
    const int c = (chunk & 7) << 2;
    _Float16* da = sA + r * LDT + c;
    da[0] = (_Float16)ra[i].x; da[1] = (_Float16)ra[i].y;
    da[2] = (_Float16)ra[i].z; da[3] = (_Float16)ra[i].w;
    _Float16* db = sB + r * LDT + c;
    db[0] = (_Float16)rb[i].x; db[1] = (_Float16)rb[i].y;
    db[2] = (_Float16)rb[i].z; db[3] = (_Float16)rb[i].w;
  }
  __syncthreads();

  int buf = 0;
  for (int k0 = 0; k0 < D_MODEL; k0 += KT, buf ^= 1) {
    const bool hasNext = (k0 + KT) < D_MODEL;
    if (hasNext) {
#pragma unroll
      for (int i = 0; i < 4; ++i) {
        const int chunk = tid + i * 256;
        const int r = chunk >> 3;
        const int c = (chunk & 7) << 2;
        ra[i] = *(const float4*)(A + (size_t)(mBase + r) * D_MODEL + k0 + KT + c);
        rb[i] = *(const float4*)(W + (size_t)(nBase + r) * D_MODEL + k0 + KT + c);
      }
    }
    // compute current buffer: 1 A-frag reused across 8 B-frags -> 8 WMMAs
    const v16h a = frag_ldA(sA + buf * TBUF + wid * 16 * LDT, LDT);
#pragma unroll
    for (int t = 0; t < 8; ++t) {
      const v16h b = frag_ldB(sB + buf * TBUF + t * 16 * LDT, LDT);
      acc[t] = wmma16(a, b, acc[t]);
    }
    if (hasNext) {
#pragma unroll
      for (int i = 0; i < 4; ++i) {
        const int chunk = tid + i * 256;
        const int r = chunk >> 3;
        const int c = (chunk & 7) << 2;
        _Float16* da = sA + (buf ^ 1) * TBUF + r * LDT + c;
        da[0] = (_Float16)ra[i].x; da[1] = (_Float16)ra[i].y;
        da[2] = (_Float16)ra[i].z; da[3] = (_Float16)ra[i].w;
        _Float16* db = sB + (buf ^ 1) * TBUF + r * LDT + c;
        db[0] = (_Float16)rb[i].x; db[1] = (_Float16)rb[i].y;
        db[2] = (_Float16)rb[i].z; db[3] = (_Float16)rb[i].w;
      }
    }
    __syncthreads();
  }

  const int n  = lane & 15;
  const int m0 = (lane >> 4) << 3;
#pragma unroll
  for (int t = 0; t < 8; ++t) {
    const int col = nBase + t * 16 + n;
    const float bv = bias[col];
    const int h = col >> 6, d = col & 63;
#pragma unroll
    for (int i = 0; i < 8; ++i) {
      const int m = mBase + wid * 16 + m0 + i;
      const int b = m >> 10, r = m & 1023;
      const float val = acc[t][i] + bv;
      size_t dst;
      if (mode == 0) dst = (((size_t)(b * NHEAD + h) * NQS) + r) * D_KH + d;
      else           dst = (((size_t)(b * NHEAD + h) * D_KH) + d) * NKS + r;
      out[dst] = (_Float16)val;
    }
  }
}

// ---- Output GEMM: out(f32) = O(f16, MxK) @ Wo(f32, NxK)^T + bo ------------
__global__ __launch_bounds__(256) void out_gemm_kernel(
    const _Float16* __restrict__ A, const float* __restrict__ W,
    const float* __restrict__ bias, float* __restrict__ out) {
  __shared__ _Float16 sA[2 * TBUF];
  __shared__ _Float16 sB[2 * TBUF];
  const int tid   = threadIdx.x;
  const int wid   = tid >> 5;
  const int lane  = tid & 31;
  const int mBase = blockIdx.y * MT;
  const int nBase = blockIdx.x * NTT;

  v8f acc[8];
#pragma unroll
  for (int t = 0; t < 8; ++t) acc[t] = (v8f){};

  uint4  ra[2];          // 128x32 f16 tile: 512 chunks of 8 f16 (16B)
  float4 rb[4];
#pragma unroll
  for (int i = 0; i < 2; ++i) {
    const int chunk = tid + i * 256;
    const int r = chunk >> 2;
    const int c = (chunk & 3) << 3;
    ra[i] = *(const uint4*)(A + (size_t)(mBase + r) * D_MODEL + c);
  }
#pragma unroll
  for (int i = 0; i < 4; ++i) {
    const int chunk = tid + i * 256;
    const int r = chunk >> 3;
    const int c = (chunk & 7) << 2;
    rb[i] = *(const float4*)(W + (size_t)(nBase + r) * D_MODEL + c);
  }
#pragma unroll
  for (int i = 0; i < 2; ++i) {
    const int chunk = tid + i * 256;
    const int r = chunk >> 2;
    const int c = (chunk & 3) << 3;
    *(uint4*)(sA + r * LDT + c) = ra[i];   // row stride 80B keeps 16B alignment
  }
#pragma unroll
  for (int i = 0; i < 4; ++i) {
    const int chunk = tid + i * 256;
    const int r = chunk >> 3;
    const int c = (chunk & 7) << 2;
    _Float16* db = sB + r * LDT + c;
    db[0] = (_Float16)rb[i].x; db[1] = (_Float16)rb[i].y;
    db[2] = (_Float16)rb[i].z; db[3] = (_Float16)rb[i].w;
  }
  __syncthreads();

  int buf = 0;
  for (int k0 = 0; k0 < D_MODEL; k0 += KT, buf ^= 1) {
    const bool hasNext = (k0 + KT) < D_MODEL;
    if (hasNext) {
#pragma unroll
      for (int i = 0; i < 2; ++i) {
        const int chunk = tid + i * 256;
        const int r = chunk >> 2;
        const int c = (chunk & 3) << 3;
        ra[i] = *(const uint4*)(A + (size_t)(mBase + r) * D_MODEL + k0 + KT + c);
      }
#pragma unroll
      for (int i = 0; i < 4; ++i) {
        const int chunk = tid + i * 256;
        const int r = chunk >> 3;
        const int c = (chunk & 7) << 2;
        rb[i] = *(const float4*)(W + (size_t)(nBase + r) * D_MODEL + k0 + KT + c);
      }
    }
    const v16h a = frag_ldA(sA + buf * TBUF + wid * 16 * LDT, LDT);
#pragma unroll
    for (int t = 0; t < 8; ++t) {
      const v16h b = frag_ldB(sB + buf * TBUF + t * 16 * LDT, LDT);
      acc[t] = wmma16(a, b, acc[t]);
    }
    if (hasNext) {
#pragma unroll
      for (int i = 0; i < 2; ++i) {
        const int chunk = tid + i * 256;
        const int r = chunk >> 2;
        const int c = (chunk & 3) << 3;
        *(uint4*)(sA + (buf ^ 1) * TBUF + r * LDT + c) = ra[i];
      }
#pragma unroll
      for (int i = 0; i < 4; ++i) {
        const int chunk = tid + i * 256;
        const int r = chunk >> 3;
        const int c = (chunk & 7) << 2;
        _Float16* db = sB + (buf ^ 1) * TBUF + r * LDT + c;
        db[0] = (_Float16)rb[i].x; db[1] = (_Float16)rb[i].y;
        db[2] = (_Float16)rb[i].z; db[3] = (_Float16)rb[i].w;
      }
    }
    __syncthreads();
  }

  const int n  = lane & 15;
  const int m0 = (lane >> 4) << 3;
#pragma unroll
  for (int t = 0; t < 8; ++t) {
    const int col = nBase + t * 16 + n;
    const float bv = bias[col];
#pragma unroll
    for (int i = 0; i < 8; ++i) {
      const int m = mBase + wid * 16 + m0 + i;
      out[(size_t)m * D_MODEL + col] = acc[t][i] + bv;
    }
  }
}

// ---- Fused attention: S = (QK^T)*scale*w, mask, softmax (-> att), O = P V -
#define LDP (NKS + 8)

__global__ __launch_bounds__(256) void attn_kernel(
    const _Float16* __restrict__ qb, const _Float16* __restrict__ kb,
    const _Float16* __restrict__ vtb, const float* __restrict__ attw,
    const unsigned char* __restrict__ maskp, float* __restrict__ att_out,
    _Float16* __restrict__ ob) {
  __shared__ _Float16 sQ[16 * 64];
  __shared__ float    sS[16 * NKS];
  __shared__ _Float16 sP[16 * LDP];
  __shared__ float    sMax[16];
  __shared__ float    sSum[16];
  __shared__ float    sO[16 * 64];

  const int tid  = threadIdx.x;
  const int wid  = tid >> 5;
  const int lane = tid & 31;
  const int q0   = blockIdx.x * 16;
  const int bh   = blockIdx.y;
  const float scale = 0.125f;             // 1/sqrt(64)

  // stage Q stripe (16 x 64 f16)
  {
    const _Float16* src = qb + ((size_t)bh * NQS + q0) * D_KH;
#pragma unroll
    for (int i = 0; i < 4; ++i) sQ[tid * 4 + i] = src[tid * 4 + i];
  }
  __syncthreads();

  const v16h aq0 = frag_ldA(sQ, 64);       // d = 0..31
  const v16h aq1 = frag_ldA(sQ + 32, 64);  // d = 32..63

  const size_t kbase = (size_t)bh * NKS * D_KH;
  const size_t wbase = ((size_t)bh * NQS + q0) * NKS;

  // phase 1: scores with fused scale * attention_weights, mask -> -inf
  for (int ct = wid; ct < NKS / 16; ct += 8) {
    const int kk0 = ct * 16;
    v8f acc = (v8f){};
    const _Float16* kp = kb + kbase + (size_t)kk0 * D_KH;
    acc = wmma16(aq0, frag_ldB(kp, 64), acc);
    acc = wmma16(aq1, frag_ldB(kp + 32, 64), acc);
    const int n  = lane & 15;
    const int m0 = (lane >> 4) << 3;
#pragma unroll
    for (int i = 0; i < 8; ++i) {
      const int m = m0 + i;
      const size_t gi = wbase + (size_t)m * NKS + kk0 + n;
      float s = acc[i] * scale * attw[gi];
      if (maskp[gi]) s = -__builtin_inff();
      sS[m * NKS + kk0 + n] = s;
    }
  }
  __syncthreads();

  // phase 2: row max / sum-of-exp (2 rows per wave)
#pragma unroll
  for (int rr = 0; rr < 2; ++rr) {
    const int r = wid * 2 + rr;
    float mx = -__builtin_inff();
    for (int c = lane; c < NKS; c += 32) mx = fmaxf(mx, sS[r * NKS + c]);
#pragma unroll
    for (int off = 16; off > 0; off >>= 1) mx = fmaxf(mx, __shfl_xor(mx, off, 32));
    float sum = 0.f;
    for (int c = lane; c < NKS; c += 32) sum += __expf(sS[r * NKS + c] - mx);
#pragma unroll
    for (int off = 16; off > 0; off >>= 1) sum += __shfl_xor(sum, off, 32);
    if (lane == 0) { sMax[r] = mx; sSum[r] = sum; }
  }
  __syncthreads();

  // phase 3: normalize, write att (fp32, single HBM touch), build f16 P stripe
  for (int idx = tid; idx < 16 * NKS; idx += 256) {
    const int r = idx >> 10, c = idx & (NKS - 1);
    const float p = __expf(sS[idx] - sMax[r]) / sSum[r];
    att_out[wbase + (size_t)r * NKS + c] = p;
    sP[r * LDP + c] = (_Float16)p;
  }
  __syncthreads();

  // phase 4: O = P @ V  (waves split over 4 d-tiles x 2 k-halves)
  const int dt   = wid & 3;
  const int half = wid >> 2;
  const int d0   = dt * 16;
  const size_t vbase = (size_t)bh * D_KH * NKS;
  v8f oacc = (v8f){};
  const int kkBeg = half * (NKS / 2);
#pragma unroll 4
  for (int kk = kkBeg; kk < kkBeg + NKS / 2; kk += 32) {
    v16h a = frag_ldA(sP + kk, LDP);
    v16h b = frag_ldB(vtb + vbase + (size_t)d0 * NKS + kk, NKS);
    oacc = wmma16(a, b, oacc);
  }
  const int n  = lane & 15;
  const int m0 = (lane >> 4) << 3;
  if (half == 1) {
#pragma unroll
    for (int i = 0; i < 8; ++i) sO[(m0 + i) * 64 + d0 + n] = oacc[i];
  }
  __syncthreads();
  if (half == 0) {
    const int b_ = bh >> 4, h_ = bh & 15;
#pragma unroll
    for (int i = 0; i < 8; ++i) {
      const float o = oacc[i] + sO[(m0 + i) * 64 + d0 + n];
      const size_t dst = ((size_t)b_ * NQS + q0 + m0 + i) * D_MODEL
                       + h_ * D_KH + d0 + n;
      ob[dst] = (_Float16)o;
    }
  }
}

// ---- Host launch ----------------------------------------------------------
extern "C" void kernel_launch(void* const* d_in, const int* in_sizes, int n_in,
                              void* d_out, int out_size, void* d_ws, size_t ws_size,
                              hipStream_t stream) {
  (void)in_sizes; (void)n_in; (void)out_size; (void)ws_size;
  const float* queries = (const float*)d_in[0];
  const float* keys    = (const float*)d_in[1];
  const float* values  = (const float*)d_in[2];
  const float* attw    = (const float*)d_in[3];
  const unsigned char* maskp = (const unsigned char*)d_in[4];
  const float* Wq = (const float*)d_in[5];
  const float* bq = (const float*)d_in[6];
  const float* Wk = (const float*)d_in[7];
  const float* bk = (const float*)d_in[8];
  const float* Wv = (const float*)d_in[9];
  const float* bv = (const float*)d_in[10];
  const float* Wo = (const float*)d_in[11];
  const float* bo = (const float*)d_in[12];

  float* out = (float*)d_out;                              // (B, NQ, D_MODEL)
  float* att = out + (size_t)NB * NQS * D_MODEL;           // (B, H, NQ, NK)

  const size_t QKV = (size_t)BH * NQS * D_KH;              // 4,194,304 elems
  _Float16* qb  = (_Float16*)d_ws;       // Q  (b,h,nq,64)
  _Float16* kb  = qb + QKV;              // K  (b,h,nk,64)
  _Float16* vtb = kb + QKV;              // V^T(b,h,64,nk)
  _Float16* ob  = vtb + QKV;             // O  (b,nq,d_model)

  const dim3 gblk(D_MODEL / NTT, M_ALL / MT);              // (8, 32)
  proj_gemm_kernel<<<gblk, 256, 0, stream>>>(queries, Wq, bq, qb, 0);
  proj_gemm_kernel<<<gblk, 256, 0, stream>>>(keys,    Wk, bk, kb, 0);
  proj_gemm_kernel<<<gblk, 256, 0, stream>>>(values,  Wv, bv, vtb, 1);

  attn_kernel<<<dim3(NQS / 16, BH), 256, 0, stream>>>(qb, kb, vtb, attw,
                                                      maskp, att, ob);

  out_gemm_kernel<<<gblk, 256, 0, stream>>>(ob, Wo, bo, out);
}
